// CrossAttentionFusionModule_33904471834922
// MI455X (gfx1250) — matrix operand order repaired and checked by reading, not compiled, not run on hardware
//
#include <hip/hip_runtime.h>

// Problem constants (reference: M=50000, K=4096, N=128, D=256, H=8, DR=6, HD=32)
#define K_SP   4096
#define N_PT   128
#define D_F    256
#define H_N    8
#define DR_N   6
#define M_SP   50000

typedef __attribute__((ext_vector_type(16))) __bf16 v16bf;
typedef __attribute__((ext_vector_type(8)))  float  v8f;

union FragAB { v16bf v; uint4 q[2]; };
union FragC  { v8f v;  float f[8]; };

#define WMMA_BF16(A,B,C) \
  __builtin_amdgcn_wmma_f32_16x16x32_bf16(false,(A),false,(B),(short)0,(C),false,false)

// ---------------- LDS layout (bytes), total < 320 KB ----------------
#define PF_OFF 0          // pf   bf16 [128][256] stride 512
#define HF_OFF 65536      // h    f32  [128][256] stride 1024 (early, dead after LN)
#define KB_OFF 65536      // kT   bf16 [256][128] stride 256  (late)
#define VB_OFF 131072     // v    bf16 [128][256] stride 512  (late)
#define QB_OFF 196608     // h_bf16 -> Q -> ctx, bf16 [128][256] stride 512
#define WK_OFF 262144     // scores f32 [64][128] s=512  /  LN staging f32 [32][256] s=1024
#define PI_OFF 294912     // point_input f32 [128][8]
#define AC_OFF 299008     // acc f32 [256]
#define RD_OFF 300032     // reduction scratch
#define SMEM_BYTES 300160

// A fragment (16x32 bf16, M x K): lane -> M, packed K per ISA 7.12.2
__device__ __forceinline__ v16bf ld_frag_a_lds(const char* base, int row0, int col0,
                                               int stride, int lane) {
  int m  = row0 + (lane & 15);
  int c0 = col0 + ((lane >> 4) << 3);       // hi half-wave starts at K+8
  const char* p = base + m * stride + c0 * 2;
  FragAB f;
  f.q[0] = *(const uint4*)(p);              // K = c0 .. c0+7
  f.q[1] = *(const uint4*)(p + 32);         // K = c0+16 .. c0+23
  return f.v;
}

// B fragment (32x16 bf16, K x N) from row-major [K][N] source: lane -> N, elems -> K (same packing as A)
__device__ __forceinline__ v16bf ld_frag_b_lds(const char* base, int k0, int n0,
                                               int stride, int lane) {
  int n  = n0 + (lane & 15);
  int kb = k0 + ((lane >> 4) << 3);
  FragAB f;
#pragma unroll
  for (int s = 0; s < 8; ++s) {
    f.v[s]     = *(const __bf16*)(base + (kb + s) * stride + n * 2);
    f.v[s + 8] = *(const __bf16*)(base + (kb + 16 + s) * stride + n * 2);
  }
  return f.v;
}

__device__ __forceinline__ v16bf ld_frag_b_glb(const __bf16* w, int k0, int n0, int lane) {
  int n  = n0 + (lane & 15);
  int kb = k0 + ((lane >> 4) << 3);
  FragAB f;
#pragma unroll
  for (int s = 0; s < 8; ++s) {
    f.v[s]     = w[(size_t)(kb + s) * D_F + n];
    f.v[s + 8] = w[(size_t)(kb + 16 + s) * D_F + n];
  }
  return f.v;
}

// C/D 16x16 f32: VGPR j -> M=row0+j (+8 for lanes 16..31), lane&15 -> N
__device__ __forceinline__ void st_c_bf16_rm(char* base, int row0, int col0, int stride,
                                             int lane, const float* c, const float* bias) {
  int n  = col0 + (lane & 15);
  int m0 = row0 + ((lane >> 4) << 3);
  float bb = bias ? bias[n] : 0.f;
#pragma unroll
  for (int j = 0; j < 8; ++j)
    *(__bf16*)(base + (m0 + j) * stride + n * 2) = (__bf16)(c[j] + bb);
}

__device__ __forceinline__ void st_c_bf16_tr(char* base, int row0, int col0, int stride,
                                             int lane, const float* c, const float* bias) {
  int n  = col0 + (lane & 15);                 // output channel -> destination row
  int m0 = row0 + ((lane >> 4) << 3);
  float bb = bias ? bias[n] : 0.f;
#pragma unroll
  for (int j = 0; j < 8; ++j)
    *(__bf16*)(base + n * stride + (m0 + j) * 2) = (__bf16)(c[j] + bb);
}

// -------------------------------------------------------------------
// Kernel 1: convert the five 256x256 f32 weights to bf16 (row-major) in d_ws.
// Region order: [0]=w_enc2 [1]=wq [2]=wk [3]=wv [4]=wo, 65536 elems each.
__global__ void convert_weights_bf16(const float* __restrict__ w0, const float* __restrict__ w1,
                                     const float* __restrict__ w2, const float* __restrict__ w3,
                                     const float* __restrict__ w4, __bf16* __restrict__ dst) {
  int i = blockIdx.x * blockDim.x + threadIdx.x;
  if (i >= 5 * 65536) return;
  int w = i >> 16, r = i & 65535;
  const float* src = (w == 0) ? w0 : (w == 1) ? w1 : (w == 2) ? w2 : (w == 3) ? w3 : w4;
  dst[i] = (__bf16)src[r];
}

// Kernel 2: base copy all_sp_features -> fused output region (scatter overwrites later).
__global__ void copy_feat(const float4* __restrict__ src, float4* __restrict__ dst, int n4) {
  int i = blockIdx.x * blockDim.x + threadIdx.x;
  int step = gridDim.x * blockDim.x;
  for (; i < n4; i += step) dst[i] = src[i];
}

// Kernel 3: one workgroup per hard superpoint.
__global__ __launch_bounds__(256, 1)
void hsp_fused_kernel(const long long* __restrict__ idxs,
                      const float* __restrict__ feat,
                      const float* __restrict__ cent,
                      const float* __restrict__ raw,
                      const float* __restrict__ w_enc1, const float* __restrict__ b_enc1,
                      const float* __restrict__ g_eln,  const float* __restrict__ b_eln,
                      const float* __restrict__ b_enc2,
                      const float* __restrict__ bq, const float* __restrict__ bk,
                      const float* __restrict__ bv, const float* __restrict__ bo,
                      const float* __restrict__ g_pn, const float* __restrict__ b_pn,
                      const float* __restrict__ g_n1, const float* __restrict__ b_n1,
                      const __bf16* __restrict__ wbf,
                      float* __restrict__ out_enh,
                      float* __restrict__ out_pf2,
                      float* __restrict__ out_fused) {
  extern __shared__ char smem[];
  const int tid  = threadIdx.x;
  const int wv   = tid >> 5;
  const int lane = tid & 31;
  const int blk  = blockIdx.x;
  const long long gi = idxs[blk];

  // ---- Phase 1: gather raw points, canonicalize vs centroid -> LDS [128][8] f32
  {
    const long long base = gi * (long long)(N_PT * DR_N);
    const float cx = cent[gi * 3 + 0], cy = cent[gi * 3 + 1], cz = cent[gi * 3 + 2];
    for (int e = tid; e < N_PT * DR_N; e += 256) {
      int p = e / DR_N, c = e % DR_N;
      float v = raw[base + e];
      if      (c == 0) v -= cx;
      else if (c == 1) v -= cy;
      else if (c == 2) v -= cz;
      *(float*)(smem + PI_OFF + (p * 8 + c) * 4) = v;
    }
  }
  __syncthreads();

  // ---- Phase 2: enc1 (6 -> 256), K-dim too small for WMMA: scalar FMA.
  // Thread t owns output channel t for all 128 points.
  {
    float wc[6];
#pragma unroll
    for (int j = 0; j < 6; ++j) wc[j] = w_enc1[j * D_F + tid];
    float bb = b_enc1[tid];
    for (int p = 0; p < N_PT; ++p) {
      const float* pi = (const float*)(smem + PI_OFF + p * 32);
      float h = bb;
#pragma unroll
      for (int j = 0; j < 6; ++j) h += pi[j] * wc[j];
      *(float*)(smem + HF_OFF + p * 1024 + tid * 4) = h;
    }
  }
  __syncthreads();

  // ---- Phase 3: LayerNorm + ReLU -> h bf16 (QB region). One row per thread.
  if (tid < N_PT) {
    const float* hr = (const float*)(smem + HF_OFF + tid * 1024);
    float s = 0.f, s2 = 0.f;
    for (int c = 0; c < D_F; ++c) { float v = hr[c]; s += v; s2 += v * v; }
    float mu = s * (1.f / D_F);
    float rs = rsqrtf(s2 * (1.f / D_F) - mu * mu + 1e-5f);
    __bf16* dst = (__bf16*)(smem + QB_OFF + tid * 512);
    for (int c = 0; c < D_F; ++c) {
      float v = (hr[c] - mu) * rs * g_eln[c] + b_eln[c];
      dst[c] = (__bf16)fmaxf(v, 0.f);
    }
  }
  __syncthreads();

  // ---- Phase 4: pf = h @ w_enc2 + b_enc2   (128x256x256, WMMA) -> PF bf16
  for (int t = wv; t < 128; t += 8) {
    int mt = t & 7, nt = t >> 3;
    FragC acc = {};
#pragma unroll
    for (int kt = 0; kt < 8; ++kt) {
      if (kt < 7) __builtin_prefetch(wbf + 0 * 65536 + (size_t)(kt + 1) * 32 * D_F + nt * 16, 0, 0);
      v16bf a = ld_frag_a_lds(smem + QB_OFF, mt * 16, kt * 32, 512, lane);
      v16bf b = ld_frag_b_glb(wbf + 0 * 65536, kt * 32, nt * 16, lane);
      acc.v = WMMA_BF16(a, b, acc.v);
    }
    st_c_bf16_rm(smem + PF_OFF, mt * 16, nt * 16, 512, lane, acc.f, b_enc2);
  }
  __syncthreads();

  // ---- Phase 5: Q,K,V projections (share A fragments). Q->QB, K^T->KB, V->VB.
  for (int t = wv; t < 128; t += 8) {
    int mt = t & 7, nt = t >> 3;
    FragC aq = {}, ak = {}, av = {};
#pragma unroll
    for (int kt = 0; kt < 8; ++kt) {
      v16bf a  = ld_frag_a_lds(smem + PF_OFF, mt * 16, kt * 32, 512, lane);
      v16bf b1 = ld_frag_b_glb(wbf + 1 * 65536, kt * 32, nt * 16, lane);
      aq.v = WMMA_BF16(a, b1, aq.v);
      v16bf b2 = ld_frag_b_glb(wbf + 2 * 65536, kt * 32, nt * 16, lane);
      ak.v = WMMA_BF16(a, b2, ak.v);
      v16bf b3 = ld_frag_b_glb(wbf + 3 * 65536, kt * 32, nt * 16, lane);
      av.v = WMMA_BF16(a, b3, av.v);
    }
    st_c_bf16_rm(smem + QB_OFF, mt * 16, nt * 16, 512, lane, aq.f, bq);
    st_c_bf16_tr(smem + KB_OFF, mt * 16, nt * 16, 256, lane, ak.f, bk);
    st_c_bf16_rm(smem + VB_OFF, mt * 16, nt * 16, 512, lane, av.f, bv);
  }
  ((float*)(smem + AC_OFF))[tid] = 0.f;   // zero super-residual accumulator
  __syncthreads();

  // ---- Phase 6: attention, per head, 64 query rows at a time.
  const float scale = 0.17677669529663687f;  // 1/sqrt(32)
  for (int h = 0; h < H_N; ++h) {
    for (int g = 0; g < 2; ++g) {
      const int qr0 = g * 64;
      // scores[64][128] = Q_h (A) x K_h^T (B), single K-chunk (HD=32) per tile
      for (int t = wv; t < 32; t += 8) {
        int mt = t & 3, nt = t >> 2;
        FragC acc = {};
        v16bf a = ld_frag_a_lds(smem + QB_OFF, qr0 + mt * 16, h * 32, 512, lane);
        v16bf b = ld_frag_b_lds(smem + KB_OFF, h * 32, nt * 16, 256, lane);
        acc.v = WMMA_BF16(a, b, acc.v);
        int n  = nt * 16 + (lane & 15);
        int m0 = mt * 16 + ((lane >> 4) << 3);
#pragma unroll
        for (int j = 0; j < 8; ++j)
          *(float*)(smem + WK_OFF + (m0 + j) * 512 + n * 4) = acc.f[j];
      }
      __syncthreads();
      // softmax: one row per thread, in-place f32 -> bf16 compaction
      if (tid < 64) {
        float* row = (float*)(smem + WK_OFF + tid * 512);
        float mx = row[0];
        for (int i = 1; i < N_PT; ++i) mx = fmaxf(mx, row[i]);
        float s = 0.f;
        for (int i = 0; i < N_PT; ++i) s += __expf((row[i] - mx) * scale);
        float inv = 1.f / s;
        __bf16* br = (__bf16*)row;
        for (int i = 0; i < N_PT; ++i) {
          float e = __expf((row[i] - mx) * scale) * inv;  // row[i] read before clobber
          br[i] = (__bf16)e;
        }
      }
      __syncthreads();
      // ctx[64][32] = attn (A, bf16 rows in WK) x V_h (B) ; overwrite dead Q_h slice
      {
        int mt = wv & 3, nt = wv >> 2;
        FragC acc = {};
#pragma unroll
        for (int kc = 0; kc < 4; ++kc) {
          v16bf a = ld_frag_a_lds(smem + WK_OFF, mt * 16, kc * 32, 512, lane);
          v16bf b = ld_frag_b_lds(smem + VB_OFF, kc * 32, h * 32 + nt * 16, 512, lane);
          acc.v = WMMA_BF16(a, b, acc.v);
        }
        st_c_bf16_rm(smem + QB_OFF, qr0 + mt * 16, h * 32 + nt * 16, 512, lane, acc.f, nullptr);
      }
      __syncthreads();
    }
  }

  // ---- Phase 7: attn_out = ctx @ wo + bo ; + pf residual ; LN(g_pn,b_pn) -> pf2
  for (int g2 = 0; g2 < 4; ++g2) {
    const int r0 = g2 * 32;
    for (int t = wv; t < 32; t += 8) {
      int mt = t & 1, nt = t >> 1;
      FragC acc = {};
#pragma unroll
      for (int kt = 0; kt < 8; ++kt) {
        if (kt < 7) __builtin_prefetch(wbf + 4 * 65536 + (size_t)(kt + 1) * 32 * D_F + nt * 16, 0, 0);
        v16bf a = ld_frag_a_lds(smem + QB_OFF, r0 + mt * 16, kt * 32, 512, lane);
        v16bf b = ld_frag_b_glb(wbf + 4 * 65536, kt * 32, nt * 16, lane);
        acc.v = WMMA_BF16(a, b, acc.v);
      }
      int n  = nt * 16 + (lane & 15);
      int m0 = mt * 16 + ((lane >> 4) << 3);
      float bb = bo[n];
#pragma unroll
      for (int j = 0; j < 8; ++j) {
        int m = m0 + j;
        float pfv = (float)*(const __bf16*)(smem + PF_OFF + (r0 + m) * 512 + n * 2);
        *(float*)(smem + WK_OFF + m * 1024 + n * 4) = acc.f[j] + bb + pfv;
      }
    }
    __syncthreads();
    if (tid < 32) {  // LayerNorm per point row, in place
      float* rowp = (float*)(smem + WK_OFF + tid * 1024);
      float s = 0.f, s2 = 0.f;
      for (int c = 0; c < D_F; ++c) { float v = rowp[c]; s += v; s2 += v * v; }
      float mu = s * (1.f / D_F);
      float rs = rsqrtf(s2 * (1.f / D_F) - mu * mu + 1e-5f);
      for (int c = 0; c < D_F; ++c) rowp[c] = (rowp[c] - mu) * rs * g_pn[c] + b_pn[c];
    }
    __syncthreads();
    // coalesced write of pf2 rows + accumulate column sums for super_residual
    float* dstg = out_pf2 + ((size_t)blk * N_PT + r0) * D_F;
    const float* stg = (const float*)(smem + WK_OFF);
    for (int i = tid; i < 32 * D_F; i += 256) dstg[i] = stg[i];
    float s = 0.f;
    for (int r = 0; r < 32; ++r) s += stg[r * D_F + tid];
    ((float*)(smem + AC_OFF))[tid] += s;   // only thread `tid` touches acc[tid]
    __syncthreads();
  }

  // ---- Phase 8: enhanced = LN(sp_feat + mean(pf2)) ; write enhanced + fused scatter
  {
    float val = feat[gi * D_F + tid] + ((const float*)(smem + AC_OFF))[tid] * (1.f / N_PT);
    float* stg = (float*)(smem + WK_OFF);
    stg[tid] = val;
    __syncthreads();
    float* red = (float*)(smem + RD_OFF);
    if (tid == 0) {
      float s = 0.f, s2 = 0.f;
      for (int c = 0; c < D_F; ++c) { float v = stg[c]; s += v; s2 += v * v; }
      float mu = s * (1.f / D_F);
      red[0] = mu;
      red[1] = s2 * (1.f / D_F) - mu * mu;
    }
    __syncthreads();
    float mu = red[0];
    float rs = rsqrtf(red[1] + 1e-5f);
    float e  = (val - mu) * rs * g_n1[tid] + b_n1[tid];
    out_enh[(size_t)blk * D_F + tid] = e;
    out_fused[(size_t)gi * D_F + tid] = e;
  }
}

// -------------------------------------------------------------------
extern "C" void kernel_launch(void* const* d_in, const int* in_sizes, int n_in,
                              void* d_out, int out_size, void* d_ws, size_t ws_size,
                              hipStream_t stream) {
  (void)in_sizes; (void)n_in; (void)out_size; (void)ws_size;
  const long long* idxs = (const long long*)d_in[0];   // int64 in reference
  const float* feat   = (const float*)d_in[1];
  const float* cent   = (const float*)d_in[2];
  const float* raw    = (const float*)d_in[3];
  const float* w_enc1 = (const float*)d_in[4];
  const float* b_enc1 = (const float*)d_in[5];
  const float* g_eln  = (const float*)d_in[6];
  const float* b_eln  = (const float*)d_in[7];
  const float* w_enc2 = (const float*)d_in[8];
  const float* b_enc2 = (const float*)d_in[9];
  const float* wq = (const float*)d_in[10]; const float* bq = (const float*)d_in[11];
  const float* wk = (const float*)d_in[12]; const float* bk = (const float*)d_in[13];
  const float* wvp= (const float*)d_in[14]; const float* bv = (const float*)d_in[15];
  const float* wo = (const float*)d_in[16]; const float* bo = (const float*)d_in[17];
  const float* g_pn = (const float*)d_in[18]; const float* b_pn = (const float*)d_in[19];
  const float* g_n1 = (const float*)d_in[20]; const float* b_n1 = (const float*)d_in[21];

  float* out       = (float*)d_out;
  float* out_enh   = out;                                   // [4096,256]
  float* out_pf2   = out + (size_t)K_SP * D_F;              // [4096,128,256]
  float* out_fused = out_pf2 + (size_t)K_SP * N_PT * D_F;   // [50000,256]
  __bf16* wbf = (__bf16*)d_ws;                              // 5 x 64K bf16 = 640 KB

  // 1) weights f32 -> bf16 (row-major [K][N]) in scratch
  convert_weights_bf16<<<(5 * 65536 + 255) / 256, 256, 0, stream>>>(
      w_enc2, wq, wk, wvp, wo, wbf);

  // 2) fused base copy: all_sp_features -> fused region (float4, coalesced)
  copy_feat<<<4096, 256, 0, stream>>>(
      (const float4*)feat, (float4*)out_fused, (M_SP * D_F) / 4);

  // 3) main fused kernel: one workgroup (8 wave32) per hard superpoint
  hsp_fused_kernel<<<K_SP, 256, SMEM_BYTES, stream>>>(
      idxs, feat, cent, raw,
      w_enc1, b_enc1, g_eln, b_eln, b_enc2,
      bq, bk, bv, bo, g_pn, b_pn, g_n1, b_n1,
      wbf, out_enh, out_pf2, out_fused);
}
